// MyBRASKModel_35845797052825
// MI455X (gfx1250) — matrix operand order repaired and checked by reading, not compiled, not run on hardware
//
#include <hip/hip_runtime.h>
#include <hip/hip_bf16.h>
#include <cstddef>

typedef __attribute__((ext_vector_type(16))) __bf16 v16bf;
typedef __attribute__((ext_vector_type(8)))  float  v8f;

static constexpr int B  = 8;
static constexpr int S  = 256;
static constexpr int H  = 768;
static constexpr int R  = 40;
static constexpr int T  = 80;
static constexpr int KS = 16;   // K_SUBJ
static constexpr int KO = 8;    // K_OBJ
#define THRS 0.5f

// ---------------------------------------------------------------------------
// Generic GEMM: Y[M,N] = X[M,K] @ W^T  (W stored [N,K] row-major, i.e. lin())
// Optional bias[N] and optional elementwise add of addX[M,N].
// One wave (32 threads) per 16x16 output tile; bf16 WMMA, f32 accumulate.
// Hot loop: unconditional float4 loads (no exec masking); guarded tail only
// for K % 32 != 0 (the tiny K=80 r_proj GEMM).
// ---------------------------------------------------------------------------
__global__ __launch_bounds__(32)
void gemm_bt_wmma(const float* __restrict__ X, const float* __restrict__ W,
                  const float* __restrict__ bias, const float* __restrict__ addX,
                  float* __restrict__ Y, int M, int N, int K)
{
  const int lane  = threadIdx.x;
  const int tileM = blockIdx.x * 16;
  const int tileN = blockIdx.y * 16;
  const int half  = lane >> 4;   // 0: lanes 0-15, 1: lanes 16-31
  const int l16   = lane & 15;

  v8f acc = {};

  // A fragment (16x32 bf16): lane<16 -> row l16, K 0..15;
  //                          lane>=16 -> row l16, K 16..31.
  // Out-of-range A rows feed only out-of-range C rows (never stored), so
  // clamp the pointer instead of predicating loads.
  const int aRowRaw = tileM + l16;
  const int aRow    = (aRowRaw < M) ? aRowRaw : (M - 1);
  const int aK0     = half * 16;
  // B fragment (32x16 bf16): lane<16 -> col l16, K 0..15; lane>=16 -> K 16..31.
  const int bCol = tileN + l16;
  const int bK0  = half * 16;

  const float* xr = X + (size_t)aRow * K;
  const float* wr = W + (size_t)bCol * K;

  const int kFull = K & ~31;
#pragma unroll 2
  for (int k0 = 0; k0 < kFull; k0 += 32) {
    v16bf a, bm;
    const float4 a0 = *(const float4*)(xr + k0 + aK0);
    const float4 a1 = *(const float4*)(xr + k0 + aK0 + 4);
    const float4 a2 = *(const float4*)(xr + k0 + aK0 + 8);
    const float4 a3 = *(const float4*)(xr + k0 + aK0 + 12);
    const float4 b0 = *(const float4*)(wr + k0 + bK0);
    const float4 b1 = *(const float4*)(wr + k0 + bK0 + 4);
    const float4 b2 = *(const float4*)(wr + k0 + bK0 + 8);
    const float4 b3 = *(const float4*)(wr + k0 + bK0 + 12);
    a[0]  = (__bf16)a0.x; a[1]  = (__bf16)a0.y; a[2]  = (__bf16)a0.z; a[3]  = (__bf16)a0.w;
    a[4]  = (__bf16)a1.x; a[5]  = (__bf16)a1.y; a[6]  = (__bf16)a1.z; a[7]  = (__bf16)a1.w;
    a[8]  = (__bf16)a2.x; a[9]  = (__bf16)a2.y; a[10] = (__bf16)a2.z; a[11] = (__bf16)a2.w;
    a[12] = (__bf16)a3.x; a[13] = (__bf16)a3.y; a[14] = (__bf16)a3.z; a[15] = (__bf16)a3.w;
    bm[0]  = (__bf16)b0.x; bm[1]  = (__bf16)b0.y; bm[2]  = (__bf16)b0.z; bm[3]  = (__bf16)b0.w;
    bm[4]  = (__bf16)b1.x; bm[5]  = (__bf16)b1.y; bm[6]  = (__bf16)b1.z; bm[7]  = (__bf16)b1.w;
    bm[8]  = (__bf16)b2.x; bm[9]  = (__bf16)b2.y; bm[10] = (__bf16)b2.z; bm[11] = (__bf16)b2.w;
    bm[12] = (__bf16)b3.x; bm[13] = (__bf16)b3.y; bm[14] = (__bf16)b3.z; bm[15] = (__bf16)b3.w;
    acc = __builtin_amdgcn_wmma_f32_16x16x32_bf16(
        false, a, false, bm, (short)0, acc, false, false);
  }

  if (kFull < K) {            // ragged tail (only K=80 GEMMs reach this)
    v16bf a, bm;
#pragma unroll
    for (int j = 0; j < 8; ++j) {
      int k1 = kFull + aK0 + j;
      int k2 = k1 + 8;
      a[j]     = (__bf16)((k1 < K) ? xr[k1] : 0.0f);
      a[j + 8] = (__bf16)((k2 < K) ? xr[k2] : 0.0f);
    }
#pragma unroll
    for (int j = 0; j < 16; ++j) {
      int kk = kFull + bK0 + j;
      bm[j] = (__bf16)((kk < K) ? wr[kk] : 0.0f);
    }
    acc = __builtin_amdgcn_wmma_f32_16x16x32_bf16(
        false, a, false, bm, (short)0, acc, false, false);
  }

  // C layout: lane<16 -> col l16, rows tileM+0..7 ; lane>=16 -> rows tileM+8..15
  const int cRow0 = tileM + half * 8;
#pragma unroll
  for (int v = 0; v < 8; ++v) {
    int rr = cRow0 + v;
    if (rr < M) {
      float o = acc[v];
      if (bias) o += bias[bCol];
      if (addX) o += addX[(size_t)rr * N + bCol];
      Y[(size_t)rr * N + bCol] = o;
    }
  }
}

// ---------------------------------------------------------------------------
// 4-way head dot products per row: o_i[row] = X[row,:] . head_w[i_i,:]
// applySig: add head_b[i] and apply sigmoid (first-level span heads).
// ---------------------------------------------------------------------------
__global__ __launch_bounds__(256)
void dot_heads(const float* __restrict__ X, const float* __restrict__ hw,
               const float* __restrict__ hb,
               int i0, int i1, int i2, int i3,
               float* o0, float* o1, float* o2, float* o3, int applySig)
{
  __shared__ float r0[256], r1[256], r2[256], r3[256];
  const int row = blockIdx.x;
  const int tid = threadIdx.x;
  const float* x = X + (size_t)row * H;
  float s0 = 0.f, s1 = 0.f, s2 = 0.f, s3 = 0.f;
  for (int h = tid; h < H; h += 256) {
    float v = x[h];
    s0 += v * hw[i0 * H + h];
    s1 += v * hw[i1 * H + h];
    s2 += v * hw[i2 * H + h];
    s3 += v * hw[i3 * H + h];
  }
  r0[tid] = s0; r1[tid] = s1; r2[tid] = s2; r3[tid] = s3;
  __syncthreads();
  for (int off = 128; off > 0; off >>= 1) {
    if (tid < off) {
      r0[tid] += r0[tid + off]; r1[tid] += r1[tid + off];
      r2[tid] += r2[tid + off]; r3[tid] += r3[tid + off];
    }
    __syncthreads();
  }
  if (tid == 0) {
    float a0 = r0[0], a1 = r1[0], a2 = r2[0], a3 = r3[0];
    if (applySig) {
      a0 = 1.f / (1.f + __expf(-(a0 + hb[i0])));
      a1 = 1.f / (1.f + __expf(-(a1 + hb[i1])));
      a2 = 1.f / (1.f + __expf(-(a2 + hb[i2])));
      a3 = 1.f / (1.f + __expf(-(a3 + hb[i3])));
    }
    if (o0) o0[row] = a0;
    if (o1) o1[row] = a1;
    if (o2) o2[row] = a2;
    if (o3) o3[row] = a3;
  }
}

// ---------------------------------------------------------------------------
// Subject-level greedy span pairing (serial per sequence, B sequences).
// ---------------------------------------------------------------------------
__global__ void greedy_subj_kernel(const float* __restrict__ sp,
                                   const float* __restrict__ ep,
                                   int* spansI, int* validI,
                                   float* spansF, float* validF)
{
  int b = threadIdx.x;
  if (b >= B) return;
  const float* s = sp + b * S;
  const float* e = ep + b * S;
  unsigned used[S / 32];
  for (int q = 0; q < S / 32; ++q) used[q] = 0u;
  int spn[KS][2]; int val[KS];
  for (int q = 0; q < KS; ++q) { spn[q][0] = 0; spn[q][1] = 0; val[q] = 0; }
  int count = 0;
  for (int i = 0; i < S; ++i) {
    if (!(s[i] > THRS)) continue;
    int j = -1;
    for (int jj = i; jj < S; ++jj) {
      if (e[jj] > THRS && !((used[jj >> 5] >> (jj & 31)) & 1u)) { j = jj; break; }
    }
    if (j < 0) continue;
    if (count < KS) { spn[count][0] = i; spn[count][1] = j; val[count] = 1; }
    used[j >> 5] |= 1u << (j & 31);
    ++count;
  }
  for (int q = 0; q < KS; ++q) {
    spansI[(b * KS + q) * 2]     = spn[q][0];
    spansI[(b * KS + q) * 2 + 1] = spn[q][1];
    validI[b * KS + q]           = val[q];
    spansF[(b * KS + q) * 2]     = (float)spn[q][0];
    spansF[(b * KS + q) * 2 + 1] = (float)spn[q][1];
    validF[b * KS + q]           = (float)val[q];
  }
}

// ---------------------------------------------------------------------------
// Object-level greedy over strided prob columns; one thread per (b,k,r).
// ---------------------------------------------------------------------------
__global__ __launch_bounds__(256)
void greedy_obj_kernel(const float* __restrict__ sp4, const float* __restrict__ ep4,
                       const int* __restrict__ subjValid,
                       float* __restrict__ spansF, float* __restrict__ validF)
{
  int t = blockIdx.x * 256 + threadIdx.x;
  const int total = B * KS * R;
  if (t >= total) return;
  int r = t % R; int bk = t / R; int k = bk % KS; int b = bk / KS;
  const size_t strideS = (size_t)KS * R;
  const size_t base = (size_t)b * S * strideS + (size_t)k * R + r;

  unsigned used[S / 32];
  for (int q = 0; q < S / 32; ++q) used[q] = 0u;
  int spn[KO][2]; int val[KO];
  for (int q = 0; q < KO; ++q) { spn[q][0] = 0; spn[q][1] = 0; val[q] = 0; }
  int count = 0;
  for (int i = 0; i < S; ++i) {
    if (!(sp4[base + (size_t)i * strideS] > THRS)) continue;
    int j = -1;
    for (int jj = i; jj < S; ++jj) {
      if (ep4[base + (size_t)jj * strideS] > THRS &&
          !((used[jj >> 5] >> (jj & 31)) & 1u)) { j = jj; break; }
    }
    if (j < 0) continue;
    if (count < KO) { spn[count][0] = i; spn[count][1] = j; val[count] = 1; }
    used[j >> 5] |= 1u << (j & 31);
    ++count;
  }
  float m = subjValid[b * KS + k] ? 1.f : 0.f;
  for (int q = 0; q < KO; ++q) {
    spansF[((size_t)t * KO + q) * 2]     = (float)spn[q][0];
    spansF[((size_t)t * KO + q) * 2 + 1] = (float)spn[q][1];
    validF[(size_t)t * KO + q]           = (float)val[q] * m;
  }
}

// ---------------------------------------------------------------------------
// s_k[b,k,:] = 0.5*(x[b,i,:] + x[b,j,:]) * valid
// ---------------------------------------------------------------------------
__global__ __launch_bounds__(256)
void gather_kernel(const float* __restrict__ x, const int* __restrict__ spansI,
                   const int* __restrict__ validI, float* __restrict__ out)
{
  int bk = blockIdx.x;
  int b  = bk / KS;
  int s0 = spansI[bk * 2], s1 = spansI[bk * 2 + 1];
  float m = validI[bk] ? 0.5f : 0.0f;
  const float* x0 = x + ((size_t)b * S + s0) * H;
  const float* x1 = x + ((size_t)b * S + s1) * H;
  float* o = out + (size_t)bk * H;
  for (int h = threadIdx.x; h < H; h += 256) o[h] = (x0[h] + x1[h]) * m;
}

__global__ __launch_bounds__(256)
void mask_rows(float* __restrict__ Y, const int* __restrict__ validI)
{
  int row = blockIdx.x;
  float m = validI[row] ? 1.f : 0.f;
  float* y = Y + (size_t)row * H;
  for (int h = threadIdx.x; h < H; h += 256) y[h] *= m;
}

// ---------------------------------------------------------------------------
// Relation-aware attention: block per (b,r); blockDim.x == S == 256.
// v[s] = tanh(tx[b,s,:] + gh[b,:] + rr[r,:]) . V_w + V_b ; A = softmax_s(v);
// C[b,r,h] = sum_s A[s] * x[b,s,h]
// ---------------------------------------------------------------------------
__global__ __launch_bounds__(256)
void attention_kernel(const float* __restrict__ tx, const float* __restrict__ gh,
                      const float* __restrict__ rrm, const float* __restrict__ x,
                      const float* __restrict__ Vw, const float* __restrict__ Vb,
                      float* __restrict__ C)
{
  __shared__ float pre[H];
  __shared__ float vwS[H];
  __shared__ float Abuf[S];
  __shared__ float red[S];
  int br = blockIdx.x; int b = br / R; int r = br % R;
  int tid = threadIdx.x;
  for (int h = tid; h < H; h += 256) {
    pre[h] = gh[b * H + h] + rrm[r * H + h];
    vwS[h] = Vw[h];
  }
  __syncthreads();

  const float* txr = tx + ((size_t)b * S + tid) * H;
  float acc = 0.f;
  for (int h = 0; h < H; h += 4) {
    float4 xv = *(const float4*)(txr + h);
    acc += tanhf(xv.x + pre[h + 0]) * vwS[h + 0];
    acc += tanhf(xv.y + pre[h + 1]) * vwS[h + 1];
    acc += tanhf(xv.z + pre[h + 2]) * vwS[h + 2];
    acc += tanhf(xv.w + pre[h + 3]) * vwS[h + 3];
  }
  acc += Vb[0];
  Abuf[tid] = acc;
  red[tid]  = acc;
  __syncthreads();
  for (int off = 128; off > 0; off >>= 1) {
    if (tid < off) red[tid] = fmaxf(red[tid], red[tid + off]);
    __syncthreads();
  }
  float mx = red[0];
  __syncthreads();
  float ex = __expf(Abuf[tid] - mx);
  red[tid] = ex;
  __syncthreads();
  for (int off = 128; off > 0; off >>= 1) {
    if (tid < off) red[tid] += red[tid + off];
    __syncthreads();
  }
  float sm = red[0];
  __syncthreads();
  Abuf[tid] = ex / sm;
  __syncthreads();

  for (int h = tid; h < H; h += 256) {
    float a2 = 0.f;
    const float* xb = x + (size_t)b * S * H + h;
    for (int s2 = 0; s2 < S; ++s2) a2 += Abuf[s2] * xb[(size_t)s2 * H];
    C[((size_t)b * R + r) * H + h] = a2;
  }
}

// ---------------------------------------------------------------------------
// probs[head,b,s,k,r] = sigmoid(t_tok + t_sub + t_rel + head_b[4+head])
// ---------------------------------------------------------------------------
__global__ __launch_bounds__(256)
void combine_probs(const float* __restrict__ tt, const float* __restrict__ ts,
                   const float* __restrict__ tr, const float* __restrict__ hb,
                   float* __restrict__ out)
{
  size_t idx = (size_t)blockIdx.x * 256 + threadIdx.x;
  const size_t total = (size_t)4 * B * S * KS * R;
  if (idx >= total) return;
  size_t t1 = idx;
  int r = t1 % R; t1 /= R;
  int k = t1 % KS; t1 /= KS;
  int s = t1 % S; t1 /= S;
  int b = t1 % B;
  int head = (int)(t1 / B);
  float v = tt[(size_t)head * B * S + b * S + s]
          + ts[head * B * KS + b * KS + k]
          + tr[head * B * R + b * R + r]
          + hb[4 + head];
  out[idx] = 1.f / (1.f + __expf(-v));
}

// ---------------------------------------------------------------------------
extern "C" void kernel_launch(void* const* d_in, const int* in_sizes, int n_in,
                              void* d_out, int out_size, void* d_ws, size_t ws_size,
                              hipStream_t stream)
{
  (void)in_sizes; (void)n_in; (void)out_size; (void)ws_size;
  const float* token    = (const float*)d_in[0];
  const float* hg       = (const float*)d_in[1];
  const float* rel_embs = (const float*)d_in[2];
  const float* b_rel_tr = (const float*)d_in[3];
  const float* head_w   = (const float*)d_in[4];
  const float* head_b   = (const float*)d_in[5];
  const float* Wmats    = (const float*)d_in[6];
  const float* Wbias    = (const float*)d_in[7];
  const float* rproj_w  = (const float*)d_in[8];
  const float* rproj_b  = (const float*)d_in[9];
  const float* V_w      = (const float*)d_in[10];
  const float* V_b      = (const float*)d_in[11];
  float* out = (float*)d_out;

  const size_t HH = (size_t)H * H;

  // ---- workspace carve ----
  float* ws = (float*)d_ws;
  size_t off = 0;
  auto alloc = [&](size_t n) { float* p = ws + off; off += n; return p; };
  float* sub_sp = alloc(B * S);
  float* sub_ep = alloc(B * S);
  float* bob_sp = alloc(B * S);
  float* bob_ep = alloc(B * S);
  float* tx     = alloc((size_t)B * S * H);
  float* txb    = alloc((size_t)B * S * H);
  float* tmix   = alloc((size_t)B * S * H);
  float* gh_f   = alloc((size_t)B * H);
  float* gh_bk  = alloc((size_t)B * H);
  float* rr_f   = alloc((size_t)R * H);
  float* rr_bk  = alloc((size_t)R * H);
  float* brel   = alloc((size_t)R * H);
  float* s_k    = alloc((size_t)B * KS * H);
  float* bs_k   = alloc((size_t)B * KS * H);
  float* s_kw   = alloc((size_t)B * KS * H);
  float* bs_kw  = alloc((size_t)B * KS * H);
  float* Cf     = alloc((size_t)B * R * H);
  float* Cb     = alloc((size_t)B * R * H);
  float* t_tok  = alloc(4 * B * S);
  float* t_sub  = alloc(4 * B * KS);
  float* t_rel  = alloc(4 * B * R);
  int* ibuf    = (int*)(ws + off);
  int* subjSpI = ibuf;                    // B*KS*2
  int* subjVI  = subjSpI + B * KS * 2;    // B*KS
  int* bSpI    = subjVI + B * KS;
  int* bVI     = bSpI + B * KS * 2;

  // ---- d_out layout (flat concat in reference return order) ----
  const size_t P1   = (size_t)B * S * KS * R;     // one prob tensor
  const size_t Ptot = 4 * P1;
  float* o_subjSp = out + Ptot;
  float* o_subjV  = o_subjSp + B * KS * 2;
  float* o_bSp    = o_subjV + B * KS;
  float* o_bV     = o_bSp + B * KS * 2;
  float* o_fSp    = o_bV + B * KS;
  float* o_fV     = o_fSp + (size_t)B * KS * R * KO * 2;
  float* o_bwSp   = o_fV + (size_t)B * KS * R * KO;
  float* o_bwV    = o_bwSp + (size_t)B * KS * R * KO * 2;

  dim3 blk32(32);
  auto gemm = [&](const float* X, const float* W, const float* bia,
                  const float* addX, float* Y, int M, int N, int K) {
    dim3 g((M + 15) / 16, N / 16);
    gemm_bt_wmma<<<g, blk32, 0, stream>>>(X, W, bia, addX, Y, M, N, K);
  };

  // 1. first-level sigmoid heads (0..3) over tokens
  dot_heads<<<B * S, 256, 0, stream>>>(token, head_w, head_b, 0, 1, 2, 3,
                                       sub_sp, sub_ep, bob_sp, bob_ep, 1);

  // 2. greedy subject spans (fwd + bwd)
  greedy_subj_kernel<<<1, 32, 0, stream>>>(sub_sp, sub_ep, subjSpI, subjVI,
                                           o_subjSp, o_subjV);
  greedy_subj_kernel<<<1, 32, 0, stream>>>(bob_sp, bob_ep, bSpI, bVI,
                                           o_bSp, o_bV);

  // 3. gather span embeddings
  gather_kernel<<<B * KS, 256, 0, stream>>>(token, subjSpI, subjVI, s_k);
  gather_kernel<<<B * KS, 256, 0, stream>>>(token, bSpI, bVI, bs_k);

  // 4. GEMMs (WMMA path)
  gemm(s_k,  Wmats + 0 * HH, Wbias + 0 * H, nullptr, s_kw,  B * KS, H, H);
  gemm(bs_k, Wmats + 0 * HH, Wbias + 0 * H, nullptr, bs_kw, B * KS, H, H);
  mask_rows<<<B * KS, 256, 0, stream>>>(s_kw, subjVI);
  mask_rows<<<B * KS, 256, 0, stream>>>(bs_kw, bVI);

  gemm(b_rel_tr, rproj_w, rproj_b, nullptr, brel, R, H, T);       // r_proj
  gemm(token,    Wmats + 3 * HH, Wbias + 3 * H, nullptr, tx,    B * S, H, H); // W_x
  gemm(hg,       Wmats + 2 * HH, Wbias + 2 * H, nullptr, gh_f,  B,     H, H); // W_g
  gemm(rel_embs, Wmats + 1 * HH, Wbias + 1 * H, nullptr, rr_f,  R,     H, H); // W_r
  gemm(token,    Wmats + 6 * HH, Wbias + 6 * H, nullptr, txb,   B * S, H, H); // W_x_b
  gemm(hg,       Wmats + 5 * HH, Wbias + 5 * H, nullptr, gh_bk, B,     H, H); // W_g_b
  gemm(brel,     Wmats + 4 * HH, Wbias + 4 * H, nullptr, rr_bk, R,     H, H); // W_r_b
  gemm(token,    Wmats + 7 * HH, Wbias + 7 * H, token,   tmix,  B * S, H, H); // x + W_x2 x

  // 5. relation-aware attention contexts
  attention_kernel<<<B * R, 256, 0, stream>>>(tx,  gh_f,  rr_f,  token, V_w, V_b, Cf);
  attention_kernel<<<B * R, 256, 0, stream>>>(txb, gh_bk, rr_bk, token, V_w, V_b, Cb);

  // 6. distributed head dots for pair_probs (heads 4..7, raw dots)
  dot_heads<<<B * S, 256, 0, stream>>>(tmix, head_w, head_b, 4, 5, 6, 7,
      t_tok, t_tok + B * S, t_tok + 2 * B * S, t_tok + 3 * B * S, 0);
  dot_heads<<<B * KS, 256, 0, stream>>>(s_kw, head_w, head_b, 4, 5, 4, 5,
      t_sub, t_sub + B * KS, nullptr, nullptr, 0);
  dot_heads<<<B * KS, 256, 0, stream>>>(bs_kw, head_w, head_b, 6, 7, 6, 7,
      t_sub + 2 * B * KS, t_sub + 3 * B * KS, nullptr, nullptr, 0);
  dot_heads<<<B * R, 256, 0, stream>>>(Cf, head_w, head_b, 4, 5, 4, 5,
      t_rel, t_rel + B * R, nullptr, nullptr, 0);
  dot_heads<<<B * R, 256, 0, stream>>>(Cb, head_w, head_b, 6, 7, 6, 7,
      t_rel + 2 * B * R, t_rel + 3 * B * R, nullptr, nullptr, 0);

  // 7. broadcast-sum sigmoid -> probs [4,B,S,KS,R] directly into d_out
  combine_probs<<<(unsigned)((Ptot + 255) / 256), 256, 0, stream>>>(
      t_tok, t_sub, t_rel, head_b, out);

  // 8. object-level greedy triples (reads probs back from d_out)
  int nObj = B * KS * R;
  greedy_obj_kernel<<<(nObj + 255) / 256, 256, 0, stream>>>(
      out + 0 * P1, out + 1 * P1, subjVI, o_fSp, o_fV);
  greedy_obj_kernel<<<(nObj + 255) / 256, 256, 0, stream>>>(
      out + 2 * P1, out + 3 * P1, bVI, o_bwSp, o_bwV);
}
